// ClipLoss_3968549782219
// MI455X (gfx1250) — matrix operand order repaired and checked by reading, not compiled, not run on hardware
//
#include <hip/hip_runtime.h>
#include <math.h>

#define N_ROWS 8192
#define DIM    512
#define TK     32

typedef __attribute__((ext_vector_type(16))) __bf16 v16bf;
typedef __attribute__((ext_vector_type(8)))  float  v8f;

union FragBF {
    unsigned int d[8];
    v16bf v;
};

// Split an fp32 value into bf16 hi + bf16 residual (bf16x3 GEMM trick).
__device__ __forceinline__ void split_bf16(float x, unsigned short& hi, unsigned short& lo) {
    unsigned int u = __float_as_uint(x);
    hi = (unsigned short)(u >> 16);                       // truncate to bf16
    float fhi = __uint_as_float(u & 0xFFFF0000u);
    float r = x - fhi;                                    // exact residual
    lo = (unsigned short)(__float_as_uint(r) >> 16);
}

// ---------------------------------------------------------------------------
// Kernel 1: zero the accumulator workspace (row_sum, col_sum, diag).
// ---------------------------------------------------------------------------
__global__ void clip_zero(float* __restrict__ p, int n) {
    int i = blockIdx.x * blockDim.x + threadIdx.x;
    if (i < n) p[i] = 0.0f;
}

// ---------------------------------------------------------------------------
// Kernel 2: diagonal logits. One wave per row, full fp32 precision.
// ---------------------------------------------------------------------------
__global__ __launch_bounds__(256)
void clip_diag(const float* __restrict__ img, const float* __restrict__ txt,
               const float* __restrict__ scale_p, float* __restrict__ diag_accum) {
    const int wave = threadIdx.x >> 5;
    const int lane = threadIdx.x & 31;
    const int row  = blockIdx.x * 8 + wave;
    const float* a = img + (size_t)row * DIM;
    const float* b = txt + (size_t)row * DIM;
    float s = 0.0f;
    #pragma unroll 4
    for (int e = lane; e < DIM; e += 32) s = fmaf(a[e], b[e], s);
    #pragma unroll
    for (int off = 1; off < 32; off <<= 1) s += __shfl_xor(s, off, 32);
    if (lane == 0) atomicAdd(diag_accum, (*scale_p) * s);
}

// ---------------------------------------------------------------------------
// Kernel 3: fused bf16x3 WMMA GEMM + streaming exp-sum accumulation.
// Block = 256 threads (8 waves), tile = 128x128, K-step = 32.
// Wave (wm in 0..3, wn in 0..1) owns a 32x64 patch = 2x4 fragments of 16x16.
// ---------------------------------------------------------------------------
__global__ __launch_bounds__(256)
void clip_gemm_lse(const float* __restrict__ img, const float* __restrict__ txt,
                   const float* __restrict__ scale_p,
                   float* __restrict__ row_sum, float* __restrict__ col_sum) {
    __shared__ unsigned short a_hi[128 * TK];
    __shared__ unsigned short a_lo[128 * TK];
    __shared__ unsigned short b_hi[128 * TK];
    __shared__ unsigned short b_lo[128 * TK];

    const int tid  = threadIdx.x;
    const int row0 = blockIdx.x * 128;
    const int col0 = blockIdx.y * 128;

    const int wave = tid >> 5;
    const int lane = tid & 31;
    const int wm   = wave >> 1;   // 0..3 : 32-row group
    const int wn   = wave & 1;    // 0..1 : 64-col group
    const int l15  = lane & 15;
    const int g    = lane >> 4;   // half-wave select

    v8f acc[2][4];
    #pragma unroll
    for (int a = 0; a < 2; ++a)
        #pragma unroll
        for (int b = 0; b < 4; ++b) acc[a][b] = {};

    // cooperative loader mapping: thread -> (row, 16-element half-row)
    const int lrow = tid >> 1;          // 0..127
    const int ks   = (tid & 1) * 16;    // 0 or 16

    for (int k0 = 0; k0 < DIM; k0 += TK) {
        // ---- load fp32 tiles, split to bf16 hi/lo, stage in LDS ----
        {
            const float4* ga = (const float4*)(img + (size_t)(row0 + lrow) * DIM + k0 + ks);
            const float4* gb = (const float4*)(txt + (size_t)(col0 + lrow) * DIM + k0 + ks);
            float av[16], bv[16];
            #pragma unroll
            for (int q = 0; q < 4; ++q) {
                float4 t = ga[q];
                av[4*q+0] = t.x; av[4*q+1] = t.y; av[4*q+2] = t.z; av[4*q+3] = t.w;
                float4 u = gb[q];
                bv[4*q+0] = u.x; bv[4*q+1] = u.y; bv[4*q+2] = u.z; bv[4*q+3] = u.w;
            }
            #pragma unroll
            for (int j = 0; j < 16; ++j) {
                unsigned short h, l;
                split_bf16(av[j], h, l);
                a_hi[lrow * TK + ks + j] = h;
                a_lo[lrow * TK + ks + j] = l;
                split_bf16(bv[j], h, l);
                b_hi[lrow * TK + ks + j] = h;
                b_lo[lrow * TK + ks + j] = l;
            }
        }
        __syncthreads();

        // ---- gather WMMA fragments from LDS ----
        // A 16x32 bf16 layout: lane -> M = lane&15; half g: elems 0..7 -> K = 8g+0..7,
        // elems 8..15 -> K = 16+8g+0..7 (consecutive K pairs => u32 reads).
        FragBF fa_h[2], fa_l[2], fb_h[4], fb_l[4];
        #pragma unroll
        for (int im = 0; im < 2; ++im) {
            const int r = wm * 32 + im * 16 + l15;
            const unsigned short* ph = &a_hi[r * TK];
            const unsigned short* pl = &a_lo[r * TK];
            #pragma unroll
            for (int j = 0; j < 4; ++j) {
                fa_h[im].d[j]     = *(const unsigned int*)(ph + 8 * g + 2 * j);
                fa_h[im].d[4 + j] = *(const unsigned int*)(ph + 16 + 8 * g + 2 * j);
                fa_l[im].d[j]     = *(const unsigned int*)(pl + 8 * g + 2 * j);
                fa_l[im].d[4 + j] = *(const unsigned int*)(pl + 16 + 8 * g + 2 * j);
            }
        }
        // B 32x16 bf16 layout: lane -> N = lane&15; half g: elems 0..7 -> K = 8g+0..7,
        // elems 8..15 -> K = 16+8g+0..7.
        #pragma unroll
        for (int in = 0; in < 4; ++in) {
            const int c = wn * 64 + in * 16 + l15;
            const unsigned short* ph = &b_hi[c * TK];
            const unsigned short* pl = &b_lo[c * TK];
            #pragma unroll
            for (int j = 0; j < 4; ++j) {
                fb_h[in].d[j]     = *(const unsigned int*)(ph + 8 * g + 2 * j);
                fb_h[in].d[4 + j] = *(const unsigned int*)(ph + 16 + 8 * g + 2 * j);
                fb_l[in].d[j]     = *(const unsigned int*)(pl + 8 * g + 2 * j);
                fb_l[in].d[4 + j] = *(const unsigned int*)(pl + 16 + 8 * g + 2 * j);
            }
        }

        // ---- bf16x3: hi*hi + hi*lo + lo*hi accumulated in fp32 ----
        #pragma unroll
        for (int im = 0; im < 2; ++im)
            #pragma unroll
            for (int in = 0; in < 4; ++in) {
                acc[im][in] = __builtin_amdgcn_wmma_f32_16x16x32_bf16(
                    false, fa_h[im].v, false, fb_h[in].v, (short)0, acc[im][in], false, false);
                acc[im][in] = __builtin_amdgcn_wmma_f32_16x16x32_bf16(
                    false, fa_h[im].v, false, fb_l[in].v, (short)0, acc[im][in], false, false);
                acc[im][in] = __builtin_amdgcn_wmma_f32_16x16x32_bf16(
                    false, fa_l[im].v, false, fb_h[in].v, (short)0, acc[im][in], false, false);
            }
        __syncthreads();
    }

    // ---- epilogue: exp(scale*dot - shift), row/col partial sums ----
    // C/D layout: VGPR r, lane -> (M = r + 8*g, N = lane&15).
    const float scale = *scale_p;
    const float shift = scale - 60.0f;   // logits <= scale (unit vectors) => args <= 60

    float cp[4] = {0.0f, 0.0f, 0.0f, 0.0f};
    #pragma unroll
    for (int im = 0; im < 2; ++im) {
        #pragma unroll
        for (int r = 0; r < 8; ++r) {
            float rs = 0.0f;
            #pragma unroll
            for (int in = 0; in < 4; ++in) {
                float v = expf(fmaf(scale, acc[im][in][r], -shift));
                rs += v;
                cp[in] += v;
            }
            // reduce across the 16-lane half (rows differ between halves)
            #pragma unroll
            for (int off = 1; off < 16; off <<= 1) rs += __shfl_xor(rs, off, 32);
            if (l15 == 0) {
                const int row = row0 + wm * 32 + im * 16 + r + 8 * g;
                atomicAdd(&row_sum[row], rs);
            }
        }
    }
    #pragma unroll
    for (int in = 0; in < 4; ++in) {
        float v = cp[in] + __shfl_xor(cp[in], 16, 32);  // lanes L and L+16 share a column
        if (g == 0) {
            const int col = col0 + wn * 64 + in * 16 + l15;
            atomicAdd(&col_sum[col], v);
        }
    }
}

// ---------------------------------------------------------------------------
// Kernel 4: final reduction -> scalar loss.
// loss = (sum_i lse_row + sum_i lse_col) / (2N) - sum_i diag / N
// ---------------------------------------------------------------------------
__global__ __launch_bounds__(256)
void clip_finalize(const float* __restrict__ row_sum, const float* __restrict__ col_sum,
                   const float* __restrict__ diag_accum, const float* __restrict__ scale_p,
                   float* __restrict__ out) {
    __shared__ float red[256];
    const int tid = threadIdx.x;
    const float shift = (*scale_p) - 60.0f;
    float local = 0.0f;
    for (int i = tid; i < N_ROWS; i += 256)
        local += (logf(row_sum[i]) + shift) + (logf(col_sum[i]) + shift);
    red[tid] = local;
    __syncthreads();
    for (int s = 128; s > 0; s >>= 1) {
        if (tid < s) red[tid] += red[tid + s];
        __syncthreads();
    }
    if (tid == 0)
        out[0] = red[0] / (2.0f * N_ROWS) - (*diag_accum) / (float)N_ROWS;
}

// ---------------------------------------------------------------------------
extern "C" void kernel_launch(void* const* d_in, const int* in_sizes, int n_in,
                              void* d_out, int out_size, void* d_ws, size_t ws_size,
                              hipStream_t stream) {
    (void)in_sizes; (void)n_in; (void)out_size; (void)ws_size;
    const float* img     = (const float*)d_in[0];
    const float* txt     = (const float*)d_in[1];
    const float* scale_p = (const float*)d_in[2];

    float* ws      = (float*)d_ws;
    float* rowsum  = ws;                 // [8192]
    float* colsum  = ws + N_ROWS;        // [8192]
    float* diag_ac = ws + 2 * N_ROWS;    // [1]

    const int nz = 2 * N_ROWS + 1;
    clip_zero<<<(nz + 255) / 256, 256, 0, stream>>>(ws, nz);
    clip_diag<<<N_ROWS / 8, 256, 0, stream>>>(img, txt, scale_p, diag_ac);
    dim3 grid(N_ROWS / 128, N_ROWS / 128);
    clip_gemm_lse<<<grid, 256, 0, stream>>>(img, txt, scale_p, rowsum, colsum);
    clip_finalize<<<1, 256, 0, stream>>>(rowsum, colsum, diag_ac, scale_p, (float*)d_out);
}
